// XinNingNetwork_11098195493532
// MI455X (gfx1250) — compile-verified, matrix operand order checked by reading, not run on hardware
//
#include <hip/hip_runtime.h>

typedef __attribute__((ext_vector_type(16))) _Float16 v16h;
typedef __attribute__((ext_vector_type(8)))  _Float16 v8h;
typedef __attribute__((ext_vector_type(8)))  float    v8f;

#define IMG_SZ 112
#define NBATCH 64
#define NLAB   68

// ---------------------------------------------------------------------------
// Weight pre-pack: fp32 HWIO w -> f16 WMMA B-fragments with channel padding.
// K-order is (tap, ci_padded): k = tap*CINP + ci, real weight exists for
// ci < CINR at w[(tap*CINR + ci)*N + n]; everything else packs as 0.
// Fragment layout: wp[((kc*ntiles + nt)*32 + lane)*16 + e] with
//   n = nt*16 + (lane&15), k = kc*32 + ((lane>>4)<<3) + e + (e>=8 ? 8 : 0)
// For dense: CINR = CINP = Ktot (tap degenerates to 0).
// ---------------------------------------------------------------------------
__global__ void pack_w_k(const float* __restrict__ w, _Float16* __restrict__ wp,
                         int Ktot, int CINR, int CINP, int N, int ntiles,
                         int total)
{
  const int i = blockIdx.x * blockDim.x + threadIdx.x;
  if (i >= total) return;
  const int e  = i & 15;
  const int l  = (i >> 4) & 31;
  const int ft = i >> 9;                 // kc*ntiles + nt
  const int nt = ft % ntiles;
  const int kc = ft / ntiles;
  const int n = nt * 16 + (l & 15);
  const int k = kc * 32 + ((l >> 4) << 3) + e + ((e >= 8) ? 8 : 0);
  const int tap = k / CINP;
  const int ci  = k - tap * CINP;
  const bool ok = (k < Ktot) && (ci < CINR) && (n < N);
  const float v = ok ? w[((size_t)tap * CINR + ci) * N + n] : 0.f;
  wp[i] = (_Float16)v;
}

__global__ void zero_page_k(_Float16* __restrict__ zp)
{
  zp[threadIdx.x] = (_Float16)0.f;
}

// ---------------------------------------------------------------------------
// Implicit-GEMM 3x3 conv on f16 activations, fused bias+relu6, f16 out.
// One wave per 16(out-pixels) x 16(out-channels) tile. CINP >= 8 and a
// multiple of 8, so each lane's two 8-half fragment runs are channel-
// contiguous -> 2x global_load_b128. OOB taps resolve to a zeroed page via
// a single pointer select per run (no per-element cndmask, no cvt).
// ---------------------------------------------------------------------------
template <int CINP, int STRIDE>
__global__ void __launch_bounds__(32) convh_wmma_k(
    const _Float16* __restrict__ in, const _Float16* __restrict__ wp,
    const float* __restrict__ bias, _Float16* __restrict__ out,
    const _Float16* __restrict__ zpage,
    int B, int H, int W, int Cout, int OH, int OW, int pad, int ntiles)
{
  const int lane  = threadIdx.x & 31;
  const int idx16 = lane & 15;
  const int khalf = (lane >> 4) << 3;
  const int mtile = blockIdx.x;
  const int ntile = blockIdx.y;

  const int M = B * OH * OW;
  const int m = mtile * 16 + idx16;
  const int mm = (m < M) ? m : 0;        // clamp; padded rows never stored
  const int b  = mm / (OH * OW);
  const int rr = mm - b * (OH * OW);
  const int oy = rr / OW;
  const int ox = rr - oy * OW;

  constexpr int K  = 9 * CINP;
  constexpr int KC = ((K + 31) & ~31) / 32;

  const _Float16* wfrag = wp + ((size_t)ntile * 32 + lane) * 16;
  const size_t wstep = (size_t)ntiles * 512;

  v8f acc = {};
#pragma unroll
  for (int kc = 0; kc < KC; ++kc) {
    const int k0 = kc * 32;
    const v16h bf = *(const v16h*)wfrag;
    wfrag += wstep;
    __builtin_prefetch((const void*)wfrag, 0, 3);

    const _Float16* pr0;
    const _Float16* pr1;
#pragma unroll
    for (int r = 0; r < 2; ++r) {
      const int kb  = k0 + khalf + r * 16;   // first K of this 8-half run
      const int tap = kb / CINP;             // run stays inside one tap
      const int ci  = kb - tap * CINP;
      const int ky  = tap / 3;
      const int kx  = tap - ky * 3;
      const int iy  = oy * STRIDE + ky - pad;
      const int ix  = ox * STRIDE + kx - pad;
      const bool ok = (tap < 9) && (unsigned)iy < (unsigned)H &&
                      (unsigned)ix < (unsigned)W;
      const _Float16* p =
          ok ? (in + ((size_t)(b * H + iy) * W + ix) * CINP + ci) : zpage;
      if (r == 0) pr0 = p; else pr1 = p;
    }
    const v8h a0 = *(const v8h*)pr0;         // K = khalf + 0..7
    const v8h a1 = *(const v8h*)pr1;         // K = khalf + 16..23
    v16h af;
#pragma unroll
    for (int e = 0; e < 8; ++e) { af[e] = a0[e]; af[e + 8] = a1[e]; }

    acc = __builtin_amdgcn_wmma_f32_16x16x32_f16(false, af, false, bf,
                                                 (short)0, acc, false, false);
  }

  // C/D layout: lanes 0-15 -> N=lane, M=v; lanes 16-31 -> N=lane-16, M=v+8
  const int n = ntile * 16 + idx16;
  if (n < Cout) {
    const float bb = bias[n];
    const int rowbase = mtile * 16 + ((lane >> 4) << 3);
#pragma unroll
    for (int v = 0; v < 8; ++v) {
      const int row = rowbase + v;
      if (row < M) {
        float y = acc[v] + bb;
        y = fminf(fmaxf(y, 0.f), 6.f);
        out[(size_t)row * Cout + n] = (_Float16)y;
      }
    }
  }
}

// ---------------------------------------------------------------------------
// Dense: f16 x[M,K] @ packed w -> fp32 out, fused bias+relu6. K % 32 == 0.
// ---------------------------------------------------------------------------
__global__ void __launch_bounds__(32) dense_wmma_k(
    const _Float16* __restrict__ xin, const _Float16* __restrict__ wp,
    const float* __restrict__ bias, float* __restrict__ out,
    int M, int N, int K, int ntiles)
{
  const int lane  = threadIdx.x & 31;
  const int idx16 = lane & 15;
  const int khalf = (lane >> 4) << 3;
  const int mtile = blockIdx.x;
  const int ntile = blockIdx.y;

  const int m = mtile * 16 + idx16;
  const int mm = (m < M) ? m : 0;

  const _Float16* wfrag = wp + ((size_t)ntile * 32 + lane) * 16;
  const size_t wstep = (size_t)ntiles * 512;
  const _Float16* arow = xin + (size_t)mm * K + khalf;

  const int KC = (K + 31) / 32;
  v8f acc = {};
  for (int kc = 0; kc < KC; ++kc) {
    const v16h bf = *(const v16h*)wfrag;
    wfrag += wstep;
    __builtin_prefetch((const void*)wfrag, 0, 3);

    const v8h a0 = *(const v8h*)(arow);
    const v8h a1 = *(const v8h*)(arow + 16);
    arow += 32;
    v16h af;
#pragma unroll
    for (int e = 0; e < 8; ++e) { af[e] = a0[e]; af[e + 8] = a1[e]; }

    acc = __builtin_amdgcn_wmma_f32_16x16x32_f16(false, af, false, bf,
                                                 (short)0, acc, false, false);
  }

  const int n = ntile * 16 + idx16;
  if (n < N) {
    const float bb = bias[n];
    const int rowbase = mtile * 16 + ((lane >> 4) << 3);
#pragma unroll
    for (int v = 0; v < 8; ++v) {
      const int row = rowbase + v;
      if (row < M) {
        float y = acc[v] + bb;
        y = fminf(fmaxf(y, 0.f), 6.f);
        out[(size_t)row * N + n] = y;
      }
    }
  }
}

// ---------------------------------------------------------------------------
// Max pool on f16, TF 'SAME' (identity -inf -> skip OOB taps).
// ---------------------------------------------------------------------------
__global__ void maxpool_k(const _Float16* __restrict__ in,
                          _Float16* __restrict__ out,
                          int B, int H, int W, int C, int OH, int OW,
                          int k, int s, int pad)
{
  const int i = blockIdx.x * blockDim.x + threadIdx.x;
  const int total = B * OH * OW * C;
  if (i >= total) return;
  int t = i;
  const int c = t % C;  t /= C;
  const int ox = t % OW; t /= OW;
  const int oy = t % OH;
  const int b = t / OH;
  float mx = -INFINITY;
  for (int dy = 0; dy < k; ++dy)
    for (int dx = 0; dx < k; ++dx) {
      const int iy = oy * s + dy - pad;
      const int ix = ox * s + dx - pad;
      if ((unsigned)iy < (unsigned)H && (unsigned)ix < (unsigned)W)
        mx = fmaxf(mx, (float)in[((size_t)(b * H + iy) * W + ix) * C + c]);
    }
  out[i] = (_Float16)mx;
}

// ---------------------------------------------------------------------------
// Global average pool (f16 in) -> f16 channel block of concat buffer
// ---------------------------------------------------------------------------
__global__ void gap_k(const _Float16* __restrict__ in,
                      _Float16* __restrict__ out,
                      int B, int HW, int C, int ostride, int ooff)
{
  const int i = blockIdx.x * blockDim.x + threadIdx.x;
  if (i >= B * C) return;
  const int c = i % C;
  const int b = i / C;
  const _Float16* p = in + (size_t)b * HW * C + c;
  float s = 0.f;
  for (int t = 0; t < HW; ++t) s += (float)p[(size_t)t * C];
  out[(size_t)b * ostride + ooff + c] = (_Float16)(s / (float)HW);
}

// ---------------------------------------------------------------------------
// heat[b,i,j] = sum_l exp(-((j-112*lx)^2 + (i-112*ly)^2))   (fp32 in/out)
// ---------------------------------------------------------------------------
__global__ void heatmap_k(const float* __restrict__ out1,
                          float* __restrict__ heat, int B)
{
  const int i = blockIdx.x * blockDim.x + threadIdx.x;
  const int total = B * IMG_SZ * IMG_SZ;
  if (i >= total) return;
  int t = i;
  const int jx = t % IMG_SZ; t /= IMG_SZ;
  const int iy = t % IMG_SZ;
  const int b  = t / IMG_SZ;
  const float* lm = out1 + (size_t)b * (2 * NLAB);
  const float fj = (float)jx, fi = (float)iy;
  float s = 0.f;
  for (int l = 0; l < NLAB; ++l) {
    const float lx = lm[2 * l]     * (float)IMG_SZ;
    const float ly = lm[2 * l + 1] * (float)IMG_SZ;
    const float dx = fj - lx;
    const float dy = fi - ly;
    s += __expf(-(dx * dx + dy * dy));
  }
  heat[i] = s;
}

// fp32 x (3ch) -> f16 8ch padded (one 16B store per pixel)
__global__ void build_x16_k(const float* __restrict__ x,
                            _Float16* __restrict__ xp, int npix)
{
  const int i = blockIdx.x * blockDim.x + threadIdx.x;
  if (i >= npix) return;
  union { _Float16 h[8]; float4 f; } u;
  u.h[0] = (_Float16)x[(size_t)i * 3 + 0];
  u.h[1] = (_Float16)x[(size_t)i * 3 + 1];
  u.h[2] = (_Float16)x[(size_t)i * 3 + 2];
  u.h[3] = (_Float16)0.f; u.h[4] = (_Float16)0.f;
  u.h[5] = (_Float16)0.f; u.h[6] = (_Float16)0.f; u.h[7] = (_Float16)0.f;
  ((float4*)xp)[i] = u.f;
}

// concat(x, heat) -> f16 8ch padded (4 real channels)
__global__ void build_x2_k(const float* __restrict__ x,
                           const float* __restrict__ heat,
                           _Float16* __restrict__ xp, int npix)
{
  const int i = blockIdx.x * blockDim.x + threadIdx.x;
  if (i >= npix) return;
  union { _Float16 h[8]; float4 f; } u;
  u.h[0] = (_Float16)x[(size_t)i * 3 + 0];
  u.h[1] = (_Float16)x[(size_t)i * 3 + 1];
  u.h[2] = (_Float16)x[(size_t)i * 3 + 2];
  u.h[3] = (_Float16)heat[i];
  u.h[4] = (_Float16)0.f; u.h[5] = (_Float16)0.f;
  u.h[6] = (_Float16)0.f; u.h[7] = (_Float16)0.f;
  ((float4*)xp)[i] = u.f;
}

__global__ void copy_k(const float* __restrict__ in, float* __restrict__ out,
                       int n)
{
  const int i = blockIdx.x * blockDim.x + threadIdx.x;
  if (i < n) out[i] = in[i];
}

// ---------------------------------------------------------------------------
// Host-side helpers
// ---------------------------------------------------------------------------
template <int CINP, int STRIDE>
static void run_conv(const _Float16* in, const _Float16* wp, const float* b,
                     _Float16* out, const _Float16* zp, int B, int H, int W,
                     int Cout, hipStream_t st)
{
  const int OH = (H + STRIDE - 1) / STRIDE;
  const int OW = (W + STRIDE - 1) / STRIDE;
  int pt = (OH - 1) * STRIDE + 3 - H;
  if (pt < 0) pt = 0;
  pt /= 2;
  const int ntiles = (Cout + 15) / 16;
  dim3 g((unsigned)((B * OH * OW + 15) / 16), (unsigned)ntiles);
  convh_wmma_k<CINP, STRIDE>
      <<<g, 32, 0, st>>>(in, wp, b, out, zp, B, H, W, Cout, OH, OW, pt, ntiles);
}

static void run_pool(const _Float16* in, _Float16* out, int B, int H, int W,
                     int C, int k, int s, hipStream_t st)
{
  const int OH = (H + s - 1) / s;
  const int OW = (W + s - 1) / s;
  int pt = (OH - 1) * s + k - H;
  if (pt < 0) pt = 0;
  pt /= 2;
  const int total = B * OH * OW * C;
  maxpool_k<<<(total + 255) / 256, 256, 0, st>>>(in, out, B, H, W, C, OH, OW,
                                                 k, s, pt);
}

static void run_gap(const _Float16* in, _Float16* out, int B, int HW, int C,
                    int ostride, int ooff, hipStream_t st)
{
  gap_k<<<(B * C + 255) / 256, 256, 0, st>>>(in, out, B, HW, C, ostride, ooff);
}

static void run_dense(const _Float16* x, const _Float16* wp, const float* b,
                      float* out, int M, int N, int K, hipStream_t st)
{
  const int ntiles = (N + 15) / 16;
  dim3 g((unsigned)((M + 15) / 16), (unsigned)ntiles);
  dense_wmma_k<<<g, 32, 0, st>>>(x, wp, b, out, M, N, K, ntiles);
}

extern "C" void kernel_launch(void* const* d_in, const int* in_sizes, int n_in,
                              void* d_out, int out_size, void* d_ws,
                              size_t ws_size, hipStream_t stream)
{
  (void)in_sizes; (void)n_in; (void)out_size; (void)ws_size;
  // setup_inputs() order: x, then params in insertion order, each (w, b)
  const float* x      = (const float*)d_in[0];
  const float* w_c11  = (const float*)d_in[1];  const float* b_c11 = (const float*)d_in[2];
  const float* w_c12  = (const float*)d_in[3];  const float* b_c12 = (const float*)d_in[4];
  const float* w_c121 = (const float*)d_in[5];  const float* b_c121= (const float*)d_in[6];
  const float* w_c13  = (const float*)d_in[7];  const float* b_c13 = (const float*)d_in[8];
  const float* w_c131 = (const float*)d_in[9];  const float* b_c131= (const float*)d_in[10];
  const float* w_c14  = (const float*)d_in[11]; const float* b_c14 = (const float*)d_in[12];
  const float* w_fc1  = (const float*)d_in[13]; const float* b_fc1 = (const float*)d_in[14];
  const float* w_c21  = (const float*)d_in[15]; const float* b_c21 = (const float*)d_in[16];
  const float* w_c22  = (const float*)d_in[17]; const float* b_c22 = (const float*)d_in[18];
  const float* w_c221 = (const float*)d_in[19]; const float* b_c221= (const float*)d_in[20];
  const float* w_c23  = (const float*)d_in[21]; const float* b_c23 = (const float*)d_in[22];
  const float* w_c231 = (const float*)d_in[23]; const float* b_c231= (const float*)d_in[24];
  const float* w_c24  = (const float*)d_in[25]; const float* b_c24 = (const float*)d_in[26];
  const float* w_fc2  = (const float*)d_in[27]; const float* b_fc2 = (const float*)d_in[28];

  float* out  = (float*)d_out;
  float* out2 = out;                            // [64,136]
  float* lmk  = out + NBATCH * 2 * NLAB;        // [64,68,2]
  float* heat = out + 2 * NBATCH * 2 * NLAB;    // [64,112,112,1]

  float* ws = (float*)d_ws;
  size_t off = 0;
  auto alloc = [&](size_t nfloats) {
    off = (off + 7) & ~(size_t)7;               // 32B align for b128/v16h
    float* p = ws + off;
    off += nfloats;
    return p;
  };
  auto allocH = [&](size_t nhalf) { return (_Float16*)alloc((nhalf + 1) / 2); };
  auto pack = [&](const float* w, int Ktot, int CINR, int CINP, int N) {
    const int nt = (N + 15) / 16;
    const int kc = ((Ktot + 31) & ~31) / 32;
    const int total = kc * nt * 512;
    _Float16* wp = allocH((size_t)total);
    pack_w_k<<<(total + 255) / 256, 256, 0, stream>>>(w, wp, Ktot, CINR, CINP,
                                                      N, nt, total);
    return wp;
  };

  const int B = NBATCH;

  // ---- zero page for OOB fragment runs ----
  _Float16* zpage = allocH(32);
  zero_page_k<<<1, 32, 0, stream>>>(zpage);

  // ---- weight packing (tiny; all up front) ----
  _Float16* p_c11  = pack(w_c11,  72,  3,  8,  16);
  _Float16* p_c12  = pack(w_c12,  144, 16, 16, 32);
  _Float16* p_c121 = pack(w_c121, 288, 32, 32, 64);
  _Float16* p_c13  = pack(w_c13,  288, 32, 32, 64);
  _Float16* p_c131 = pack(w_c131, 576, 64, 64, 64);
  _Float16* p_c14  = pack(w_c14,  576, 64, 64, 64);
  _Float16* p_fc1  = pack(w_fc1,  192, 192, 192, 136);
  _Float16* p_c21  = pack(w_c21,  72,  4,  8,  8);
  _Float16* p_c22  = pack(w_c22,  72,  8,  8,  16);
  _Float16* p_c221 = pack(w_c221, 144, 16, 16, 64);
  _Float16* p_c23  = pack(w_c23,  144, 16, 16, 64);
  _Float16* p_c231 = pack(w_c231, 576, 64, 64, 64);
  _Float16* p_c24  = pack(w_c24,  576, 64, 64, 64);
  _Float16* p_fc2  = pack(w_fc2,  192, 192, 192, 136);

  const int npix = B * IMG_SZ * IMG_SZ;

  // ---- stage 1 (all activations f16) ----
  _Float16* x16 = allocH((size_t)npix * 8);
  build_x16_k<<<(npix + 255) / 256, 256, 0, stream>>>(x, x16, npix);

  _Float16* c11 = allocH((size_t)B * 56 * 56 * 16);
  run_conv<8, 2>(x16, p_c11, b_c11, c11, zpage, B, 112, 112, 16, stream);
  _Float16* c12 = allocH((size_t)B * 28 * 28 * 32);
  run_conv<16, 2>(c11, p_c12, b_c12, c12, zpage, B, 56, 56, 32, stream);
  _Float16* p12 = allocH((size_t)B * 14 * 14 * 32);
  run_pool(c12, p12, B, 28, 28, 32, 2, 2, stream);

  _Float16* fcin1 = allocH((size_t)B * 192);
  _Float16* t121 = allocH((size_t)B * 7 * 7 * 64);
  run_conv<32, 2>(p12, p_c121, b_c121, t121, zpage, B, 14, 14, 64, stream);
  run_gap(t121, fcin1, B, 49, 64, 192, 0, stream);

  _Float16* c13 = allocH((size_t)B * 7 * 7 * 64);
  run_conv<32, 2>(p12, p_c13, b_c13, c13, zpage, B, 14, 14, 64, stream);
  _Float16* p13 = allocH((size_t)B * 4 * 4 * 64);
  run_pool(c13, p13, B, 7, 7, 64, 2, 2, stream);

  _Float16* t131 = allocH((size_t)B * 2 * 2 * 64);
  run_conv<64, 2>(p13, p_c131, b_c131, t131, zpage, B, 4, 4, 64, stream);
  run_gap(t131, fcin1, B, 4, 64, 192, 64, stream);

  _Float16* t141 = allocH((size_t)B * 2 * 2 * 64);
  run_conv<64, 2>(p13, p_c14, b_c14, t141, zpage, B, 4, 4, 64, stream);
  run_gap(t141, fcin1, B, 4, 64, 192, 128, stream);

  float* out1 = alloc((size_t)B * 2 * NLAB);
  run_dense(fcin1, p_fc1, b_fc1, out1, B, 2 * NLAB, 192, stream);

  heatmap_k<<<(npix + 255) / 256, 256, 0, stream>>>(out1, heat, B);
  copy_k<<<(B * 2 * NLAB + 255) / 256, 256, 0, stream>>>(out1, lmk,
                                                         B * 2 * NLAB);

  // ---- stage 2 ----
  _Float16* x2 = allocH((size_t)npix * 8);
  build_x2_k<<<(npix + 255) / 256, 256, 0, stream>>>(x, heat, x2, npix);

  _Float16* c21 = allocH((size_t)B * 56 * 56 * 8);
  run_conv<8, 2>(x2, p_c21, b_c21, c21, zpage, B, 112, 112, 8, stream);
  _Float16* p21 = allocH((size_t)B * 28 * 28 * 8);
  run_pool(c21, p21, B, 56, 56, 8, 3, 2, stream);

  _Float16* c22 = allocH((size_t)B * 28 * 28 * 16);
  run_conv<8, 1>(p21, p_c22, b_c22, c22, zpage, B, 28, 28, 16, stream);
  _Float16* p22 = allocH((size_t)B * 14 * 14 * 16);
  run_pool(c22, p22, B, 28, 28, 16, 3, 2, stream);

  _Float16* fcin2 = allocH((size_t)B * 192);
  _Float16* t221 = allocH((size_t)B * 14 * 14 * 64);
  run_conv<16, 1>(p22, p_c221, b_c221, t221, zpage, B, 14, 14, 64, stream);
  run_gap(t221, fcin2, B, 196, 64, 192, 0, stream);

  _Float16* c23 = allocH((size_t)B * 7 * 7 * 64);
  run_conv<16, 2>(p22, p_c23, b_c23, c23, zpage, B, 14, 14, 64, stream);
  _Float16* p23 = allocH((size_t)B * 4 * 4 * 64);
  run_pool(c23, p23, B, 7, 7, 64, 3, 2, stream);

  _Float16* t231 = allocH((size_t)B * 4 * 4 * 64);
  run_conv<64, 1>(p23, p_c231, b_c231, t231, zpage, B, 4, 4, 64, stream);
  run_gap(t231, fcin2, B, 16, 64, 192, 64, stream);

  _Float16* t241 = allocH((size_t)B * 2 * 2 * 64);
  run_conv<64, 2>(p23, p_c24, b_c24, t241, zpage, B, 4, 4, 64, stream);
  run_gap(t241, fcin2, B, 4, 64, 192, 128, stream);

  run_dense(fcin2, p_fc2, b_fc2, out2, B, 2 * NLAB, 192, stream);
}
